// Repro_89850715832656
// MI455X (gfx1250) — matrix operand order repaired and checked by reading, not compiled, not run on hardware
//
#include <hip/hip_runtime.h>

typedef unsigned short u16;
typedef __attribute__((ext_vector_type(16))) __bf16 v16bf;
typedef __attribute__((ext_vector_type(8)))  float  v8f;

#define Bsz 16
#define Ssz 512
#define Dsz 256
#define Hsz 8
#define HDsz 32
#define Fsz 682
#define Msz (Bsz*Ssz)          // 8192 rows
#define NQKV 576               // 256 q + 32 k + 256 v, padded to 64-mult
#define FP 704                 // F padded to 32/64-mult
#define NGU (2*FP)             // 1408

static __device__ __forceinline__ u16 f2bf(float f) {
    unsigned int u = __float_as_uint(f);
    u = (u + 0x7FFFu + ((u >> 16) & 1u)) >> 16;
    return (u16)u;
}

// 16 bf16 elements from two 16-byte chunks (A-fragment halves)
static __device__ __forceinline__ v16bf ld_frag(const u16* p0, const u16* p1) {
    union { v16bf v; int4 q[2]; } u;
    u.q[0] = *(const int4*)p0;
    u.q[1] = *(const int4*)p1;
    return u.v;
}
// 16 bf16 elements, contiguous 32 bytes (B-fragment per-lane slice)
static __device__ __forceinline__ v16bf ld_frag2(const u16* p) {
    union { v16bf v; int4 q[2]; } u;
    u.q[0] = ((const int4*)p)[0];
    u.q[1] = ((const int4*)p)[1];
    return u.v;
}

static __device__ __forceinline__ v8f wmma_bf16(v16bf a, v16bf b, v8f c) {
    return __builtin_amdgcn_wmma_f32_16x16x32_bf16(false, a, false, b, (short)0, c, false, false);
}

// ---------------- packing ----------------
__global__ void pack_w(const float* __restrict__ src, u16* __restrict__ dst,
                       int Nvalid, int Kvalid, int Kpad, int Npad) {
    long idx = (long)blockIdx.x * 256 + threadIdx.x;
    long total = (long)Npad * Kpad;
    if (idx >= total) return;
    int n = (int)(idx / Kpad), k = (int)(idx % Kpad);
    float v = (n < Nvalid && k < Kvalid) ? src[(long)n * Kvalid + k] : 0.0f;
    dst[idx] = f2bf(v);
}

__global__ void pack_b(const float* __restrict__ src, float* __restrict__ dst,
                       int Nvalid, int Npad) {
    int i = blockIdx.x * 256 + threadIdx.x;
    if (i < Npad) dst[i] = (i < Nvalid) ? src[i] : 0.0f;
}

// ---------------- embedding + pos ----------------
__global__ void embed_kernel(const float* __restrict__ emb, const float* __restrict__ pos,
                             const int* __restrict__ tok, float* __restrict__ h) {
    int row = blockIdx.x;          // b*S + s
    int tid = threadIdx.x;         // 0..255
    int s = row & (Ssz - 1);
    int t = tok[row];
    h[(long)row * Dsz + tid] = emb[(long)t * Dsz + tid] + pos[(long)s * Dsz + tid];
}

// ---------------- rmsnorm (f32 reduce, bf16 + optional f32 out) ----------------
__global__ void rmsnorm_kernel(const float* __restrict__ in, const float* __restrict__ w,
                               u16* __restrict__ out_bf, float* __restrict__ out_f32) {
    int row = blockIdx.x, tid = threadIdx.x;    // 256 threads
    float v = in[(long)row * Dsz + tid];
    __shared__ float red[256];
    red[tid] = v * v;
    __syncthreads();
    for (int s = 128; s > 0; s >>= 1) {
        if (tid < s) red[tid] += red[tid + s];
        __syncthreads();
    }
    float rms = rsqrtf(red[0] * (1.0f / Dsz) + 1.1920928955078125e-07f);
    float x = v * rms * w[tid];
    out_bf[(long)row * Dsz + tid] = f2bf(x);
    if (out_f32) out_f32[(long)row * Dsz + tid] = x;
}

// ---------------- generic WMMA GEMM: out = A(bf16,MxK) * W(bf16,NxK)^T + bias (+resid) ----------------
// Register double-buffered K loop: iteration k+1's fragments are issued before
// iteration k's WMMAs consume their buffers, so waits become partial (overlap).
__global__ __launch_bounds__(128)
void gemm_kernel(const u16* __restrict__ A, const u16* __restrict__ Bw,
                 const float* __restrict__ bias, const float* __restrict__ resid,
                 float* __restrict__ out, int K, int N) {
    int lane = threadIdx.x, wave = threadIdx.y;
    int half = lane >> 4, l16 = lane & 15;
    int mt  = blockIdx.y * 4 + wave;   // 16-row tile
    int nt0 = blockIdx.x * 4;          // 4 x 16-col tiles
    v8f acc[4] = {};
    const u16* Arow = A + ((long)mt * 16 + l16) * K;
    const u16* Bb   = Bw + ((long)nt0 * 16 + l16) * K + half * 16;
    const long bstep = (long)16 * K;   // per 16-col tile

    v16bf a  = ld_frag(Arow + half * 16, Arow + half * 16 + 8);
    v16bf b0 = ld_frag2(Bb);
    v16bf b1 = ld_frag2(Bb + bstep);
    v16bf b2 = ld_frag2(Bb + 2 * bstep);
    v16bf b3 = ld_frag2(Bb + 3 * bstep);

    for (int k0 = 0; k0 < K; k0 += 32) {
        v16bf an = a, bn0 = b0, bn1 = b1, bn2 = b2, bn3 = b3;
        int k1 = k0 + 32;
        if (k1 < K) {
            an  = ld_frag(Arow + k1 + half * 16, Arow + k1 + half * 16 + 8);
            bn0 = ld_frag2(Bb + k1);
            bn1 = ld_frag2(Bb + bstep + k1);
            bn2 = ld_frag2(Bb + 2 * bstep + k1);
            bn3 = ld_frag2(Bb + 3 * bstep + k1);
        }
        acc[0] = wmma_bf16(a, b0, acc[0]);
        acc[1] = wmma_bf16(a, b1, acc[1]);
        acc[2] = wmma_bf16(a, b2, acc[2]);
        acc[3] = wmma_bf16(a, b3, acc[3]);
        a = an; b0 = bn0; b1 = bn1; b2 = bn2; b3 = bn3;
    }
#pragma unroll
    for (int j = 0; j < 4; ++j) {
        int col = (nt0 + j) * 16 + l16;
        float bv = bias ? bias[col] : 0.0f;
#pragma unroll
        for (int i = 0; i < 8; ++i) {
            long row = (long)mt * 16 + i + half * 8;
            float v = acc[j][i] + bv;
            if (resid) v += resid[row * N + col];
            out[row * N + col] = v;
        }
    }
}

// ---------------- RoPE + layout: qkv(f32,Mx576) -> qb[b,h,s,d], kb[b,s,d], vT[b,h,d,t] ----------------
__global__ void rope_kernel(const float* __restrict__ qkv,
                            const float* __restrict__ cosT, const float* __restrict__ sinT,
                            u16* __restrict__ qb, u16* __restrict__ kb, u16* __restrict__ vT) {
    int row = blockIdx.x;          // b*S + s
    int t = threadIdx.x;           // 0..127
    int b = row >> 9, s = row & (Ssz - 1);
    const float* r = qkv + (long)row * NQKV;
    {   // q: pair p = t (covers D=256)
        int p = t;
        float e = r[2 * p], o = r[2 * p + 1];
        int head = p >> 4, pin = p & 15;
        float c  = cosT[s * HDsz + 2 * pin];
        float sn = sinT[s * HDsz + 2 * pin];
        long base = (((long)b * Hsz + head) * Ssz + s) * HDsz + 2 * pin;
        qb[base]     = f2bf(e * c - o * sn);
        qb[base + 1] = f2bf(e * sn + o * c);
    }
    if (t < 16) {   // k: HD=32 -> 16 pairs
        float e = r[Dsz + 2 * t], o = r[Dsz + 2 * t + 1];
        float c = cosT[s * HDsz + 2 * t], sn = sinT[s * HDsz + 2 * t];
        long base = ((long)b * Ssz + s) * HDsz + 2 * t;
        kb[base]     = f2bf(e * c - o * sn);
        kb[base + 1] = f2bf(e * sn + o * c);
    }
    for (int d = t; d < Dsz; d += 128) {   // v transpose (no rope)
        float val = r[Dsz + HDsz + d];
        int head = d >> 5, dd = d & 31;
        vT[(((long)b * Hsz + head) * HDsz + dd) * Ssz + s] = f2bf(val);
    }
}

// ---------------- fused causal attention, one wave per (b, h, 16-row tile); h2 = softmax(qk^T/sqrt(hd))v + h
__global__ __launch_bounds__(32)
void attn_kernel(const u16* __restrict__ qb, const u16* __restrict__ kb,
                 const u16* __restrict__ vT, const float* __restrict__ h,
                 float* __restrict__ h2) {
    __shared__ __align__(16) float scoresF[16 * Ssz];
    __shared__ __align__(16) u16   probsH[16 * Ssz];
    int sT = blockIdx.x, hd = blockIdx.y, b = blockIdx.z;
    int lane = threadIdx.x, half = lane >> 4, l16 = lane & 15;
    int s0 = sT * 16;

    for (int i = lane; i < 16 * Ssz; i += 32) probsH[i] = 0;   // bf16 zero = 0x0000

    // Q A-fragment (HD=32 = exactly one K chunk), held for the whole tile
    const u16* qrow = qb + (((long)b * Hsz + hd) * Ssz + s0 + l16) * HDsz;
    v16bf aq = ld_frag(qrow + half * 16, qrow + half * 16 + 8);

    const float scale = 0.17677669529663687f;  // 1/sqrt(32)
    const u16* kb_base = kb + ((long)b * Ssz + l16) * HDsz + half * 16;
    v16bf bk_cur = ld_frag2(kb_base);          // tile 0
    for (int tT = 0; tT <= sT; ++tT) {         // causal: skip upper tiles entirely
        v16bf bk_nxt = bk_cur;
        if (tT < sT) bk_nxt = ld_frag2(kb_base + (long)(tT + 1) * 16 * HDsz);
        v8f c = {};
        c = wmma_bf16(aq, bk_cur, c);
        int colbase = tT * 16 + l16;
#pragma unroll
        for (int i = 0; i < 8; ++i)
            scoresF[(i + half * 8) * Ssz + colbase] = c[i] * scale;
        bk_cur = bk_nxt;
    }
    __syncthreads();

    {   // paired-lane softmax: lanes {r, r+16} cooperate on row r (stride-2 scan)
        int r = l16, count = s0 + r + 1;
        float* rowp = scoresF + r * Ssz;
        float m = -3.0e38f;
        for (int t = half; t < count; t += 2) m = fmaxf(m, rowp[t]);
        m = fmaxf(m, __shfl_xor(m, 16));
        float sum = 0.0f;
        for (int t = half; t < count; t += 2) {
            float e = __expf(rowp[t] - m);
            rowp[t] = e; sum += e;
        }
        sum += __shfl_xor(sum, 16);
        float inv = 1.0f / sum;
        u16* prow = probsH + r * Ssz;
        for (int t = half; t < count; t += 2) prow[t] = f2bf(rowp[t] * inv);
    }
    __syncthreads();

    v8f acc0 = {}, acc1 = {};
    int nChunks = (sT + 2) >> 1;   // ceil((sT+1)*16 / 32); pad probs are zero
    const u16* vbase  = vT + (((long)b * Hsz + hd) * HDsz) * Ssz;
    const u16* pbase  = probsH + l16 * Ssz + half * 16;
    const u16* v0base = vbase + (long)l16 * Ssz + half * 16;
    const u16* v1base = vbase + (long)(16 + l16) * Ssz + half * 16;
    v16bf ap  = ld_frag(pbase, pbase + 8);
    v16bf bv0 = ld_frag2(v0base);
    v16bf bv1 = ld_frag2(v1base);
    for (int kc = 0; kc < nChunks; ++kc) {
        v16bf apn = ap, bv0n = bv0, bv1n = bv1;
        if (kc + 1 < nChunks) {
            int t1 = (kc + 1) * 32;
            apn  = ld_frag(pbase + t1, pbase + t1 + 8);
            bv0n = ld_frag2(v0base + t1);
            bv1n = ld_frag2(v1base + t1);
        }
        acc0 = wmma_bf16(ap, bv0, acc0);
        acc1 = wmma_bf16(ap, bv1, acc1);
        ap = apn; bv0 = bv0n; bv1 = bv1n;
    }
#pragma unroll
    for (int i = 0; i < 8; ++i) {   // h2 = o + h  (heads write disjoint 32-col slices)
        int srow = s0 + i + half * 8;
        long base = ((long)b * Ssz + srow) * Dsz + hd * HDsz;
        h2[base + l16]      = acc0[i] + h[base + l16];
        h2[base + 16 + l16] = acc1[i] + h[base + 16 + l16];
    }
}

// ---------------- SiLU(g) * up -> padded bf16 ----------------
__global__ void silu_mul(const float* __restrict__ guw, u16* __restrict__ gu) {
    long idx = (long)blockIdx.x * 256 + threadIdx.x;
    if (idx >= (long)Msz * FP) return;
    long m = idx / FP; int f = (int)(idx % FP);
    float v = 0.0f;
    if (f < Fsz) {
        float g = guw[m * NGU + f];
        float u = guw[m * NGU + FP + f];
        v = g / (1.0f + __expf(-g)) * u;
    }
    gu[idx] = f2bf(v);
}

extern "C" void kernel_launch(void* const* d_in, const int* in_sizes, int n_in,
                              void* d_out, int out_size, void* d_ws, size_t ws_size,
                              hipStream_t stream) {
    (void)in_sizes; (void)n_in; (void)out_size; (void)ws_size;
    const float* emb        = (const float*)d_in[0];
    const float* pos_emb    = (const float*)d_in[1];
    const float* attn_nw    = (const float*)d_in[2];
    const float* wq         = (const float*)d_in[3];
    const float* bq         = (const float*)d_in[4];
    const float* wk         = (const float*)d_in[5];
    const float* bk         = (const float*)d_in[6];
    const float* wv         = (const float*)d_in[7];
    const float* bv         = (const float*)d_in[8];
    const float* cos_tab    = (const float*)d_in[9];
    const float* sin_tab    = (const float*)d_in[10];
    const float* ffn_nw     = (const float*)d_in[11];
    const float* w_gate     = (const float*)d_in[12];
    const float* b_gate     = (const float*)d_in[13];
    const float* w_up       = (const float*)d_in[14];
    const float* b_up       = (const float*)d_in[15];
    const float* w_down     = (const float*)d_in[16];
    const float* b_down     = (const float*)d_in[17];
    const int*   tok        = (const int*)d_in[18];
    // d_in[19] = mask: causal handled analytically

    char* ws = (char*)d_ws;
    size_t off = 0;
    auto alloc = [&](size_t bytes) -> char* {
        char* p = ws + off; off += (bytes + 255) & ~(size_t)255; return p;
    };
    float* h    = (float*)alloc((size_t)Msz * Dsz * 4);
    float* h2   = (float*)alloc((size_t)Msz * Dsz * 4);
    float* x2   = (float*)alloc((size_t)Msz * Dsz * 4);
    u16*   xb   = (u16*)  alloc((size_t)Msz * Dsz * 2);
    u16*   x2b  = (u16*)  alloc((size_t)Msz * Dsz * 2);
    u16*   qbuf = (u16*)  alloc((size_t)Bsz * Hsz * Ssz * HDsz * 2);
    u16*   kbuf = (u16*)  alloc((size_t)Bsz * Ssz * HDsz * 2);
    u16*   vT   = (u16*)  alloc((size_t)Bsz * Hsz * Ssz * HDsz * 2);
    u16*   wqkv = (u16*)  alloc((size_t)NQKV * Dsz * 2);
    float* bqkv = (float*)alloc((size_t)NQKV * 4);
    u16*   wgu  = (u16*)  alloc((size_t)NGU * Dsz * 2);
    float* bgu  = (float*)alloc((size_t)NGU * 4);
    u16*   wdn  = (u16*)  alloc((size_t)Dsz * FP * 2);
    u16*   gu   = (u16*)  alloc((size_t)Msz * FP * 2);
    float* guw  = (float*)alloc((size_t)Msz * NGU * 4);
    float* qkv  = guw;   // overlap: qkv (Mx576 f32) consumed before guw is written

    auto blks = [](long n) { return (unsigned)((n + 255) / 256); };

    // weight/bias packing (f32 -> padded bf16)
    pack_w<<<blks(256L * 256), 256, 0, stream>>>(wq,     wqkv,             256, 256, 256, 256);
    pack_w<<<blks(32L  * 256), 256, 0, stream>>>(wk,     wqkv + 256 * 256,  32, 256, 256, 32);
    pack_w<<<blks(288L * 256), 256, 0, stream>>>(wv,     wqkv + 288 * 256, 256, 256, 256, 288);
    pack_w<<<blks((long)FP * 256), 256, 0, stream>>>(w_gate, wgu,            Fsz, 256, 256, FP);
    pack_w<<<blks((long)FP * 256), 256, 0, stream>>>(w_up,   wgu + FP * 256, Fsz, 256, 256, FP);
    pack_w<<<blks(256L * FP), 256, 0, stream>>>(w_down, wdn,              256, Fsz, FP, 256);
    pack_b<<<blks(256), 256, 0, stream>>>(bq,     bqkv,       256, 256);
    pack_b<<<blks(32),  256, 0, stream>>>(bk,     bqkv + 256,  32, 32);
    pack_b<<<blks(288), 256, 0, stream>>>(bv,     bqkv + 288, 256, 288);
    pack_b<<<blks(FP),  256, 0, stream>>>(b_gate, bgu,        Fsz, FP);
    pack_b<<<blks(FP),  256, 0, stream>>>(b_up,   bgu + FP,   Fsz, FP);

    // h = emb[tok] + pos ; x = rmsnorm(h)
    embed_kernel  <<<Msz, 256, 0, stream>>>(emb, pos_emb, tok, h);
    rmsnorm_kernel<<<Msz, 256, 0, stream>>>(h, attn_nw, xb, nullptr);

    // fused QKV GEMM: [q|k|v] = x @ [wq;wk;wv]^T + bias
    gemm_kernel<<<dim3(NQKV / 64, Msz / 64), dim3(32, 4), 0, stream>>>(
        xb, wqkv, bqkv, nullptr, qkv, Dsz, NQKV);

    // RoPE + head layouts
    rope_kernel<<<Msz, 128, 0, stream>>>(qkv, cos_tab, sin_tab, qbuf, kbuf, vT);

    // fused causal attention + residual: h2 = attn(q,k,v) + h
    attn_kernel<<<dim3(Ssz / 16, Hsz, Bsz), 32, 0, stream>>>(qbuf, kbuf, vT, h, h2);

    // x2 = rmsnorm(h2)
    rmsnorm_kernel<<<Msz, 256, 0, stream>>>(h2, ffn_nw, x2b, x2);

    // fused gate/up GEMM, SiLU*up, down GEMM (+ b_down + x2) -> d_out
    gemm_kernel<<<dim3(NGU / 64, Msz / 64), dim3(32, 4), 0, stream>>>(
        x2b, wgu, bgu, nullptr, guw, Dsz, NGU);
    silu_mul<<<blks((long)Msz * FP), 256, 0, stream>>>(guw, gu);
    gemm_kernel<<<dim3(Dsz / 64, Msz / 64), dim3(32, 4), 0, stream>>>(
        gu, wdn, b_down, x2, (float*)d_out, FP, Dsz);
}
